// Attention_72103910965510
// MI455X (gfx1250) — compile-verified, matrix operand order checked by reading, not run on hardware
//
#include <hip/hip_runtime.h>
#include <hip/hip_bf16.h>
#include <math.h>

typedef __attribute__((ext_vector_type(16))) _Float16 v16h;
typedef __attribute__((ext_vector_type(8)))  _Float16 v8h;
typedef __attribute__((ext_vector_type(8)))  float    v8f;
typedef __attribute__((ext_vector_type(4)))  int      v4i;

// Address-space qualified element types for the async global->LDS builtin:
// param0: v4i in AS1 (global, prints as "__device__"), param1: v4i in AS3 (LDS).
typedef __attribute__((address_space(1))) v4i gv4i;
typedef __attribute__((address_space(3))) v4i lv4i;

#define NB 16
#define NT 4096
#define ND 512
#define NROWS (NB * NT)          // 65536 flattened (b,t) rows
#define LDSLD (ND + 8)           // padded LDS row stride (halfs) -> conflict-free

// Async global->LDS path if the toolchain exposes the gfx1250 builtins.
#if defined(__HIP_DEVICE_COMPILE__) && defined(__gfx1250__) && \
    __has_builtin(__builtin_amdgcn_global_load_async_to_lds_b128) && \
    __has_builtin(__builtin_amdgcn_s_wait_asynccnt)
#define USE_ASYNC_LDS 1
#else
#define USE_ASYNC_LDS 0
#endif

// Branch-free tanh: 1 - 2/(exp(2y)+1); saturates to +/-1, no EXEC divergence.
__device__ __forceinline__ float fast_tanh(float y) {
    float e = __expf(2.0f * y);
    return 1.0f - 2.0f * __builtin_amdgcn_rcpf(e + 1.0f);
}

// ---------------------------------------------------------------------------
// Kernel 1: Wt[n][k] = (f16) W[k][n]   (transpose + convert, 512 KB, L2-hot)
// ---------------------------------------------------------------------------
__global__ void __launch_bounds__(256)
prep_wt_kernel(const float* __restrict__ W, _Float16* __restrict__ Wt) {
    int idx = blockIdx.x * 256 + threadIdx.x;   // 0 .. 512*512-1
    int n = idx & (ND - 1);
    int k = idx >> 9;
    Wt[(size_t)n * ND + k] = (_Float16)W[(size_t)k * ND + n];
}

// ---------------------------------------------------------------------------
// Kernel 2: scores[row] = tanh(x[row,:] @ W) . v
// Block = 256 threads = 8 waves, covers 256 rows (each wave: 2 M-tiles).
// B (Wt) chunk of 64 cols x 512 K staged in LDS, shared by all 8 waves.
// ---------------------------------------------------------------------------
__global__ void __launch_bounds__(256)
score_kernel(const float* __restrict__ x, const _Float16* __restrict__ Wt,
             const float* __restrict__ attv, float* __restrict__ scores) {
    __shared__ _Float16 Bsh[64 * LDSLD];        // 65 KB, padded stride

    const int tid  = threadIdx.x;
    const int lane = tid & 31;
    const int wave = tid >> 5;
    const int hs   = lane >> 4;                 // half-wave select
    const int l15  = lane & 15;

    const int mt0  = blockIdx.x * 16 + wave * 2;  // first of 2 M-tiles
    const int row0 = mt0 * 16;                    // rows row0 .. row0+31

    const float* xr[2];
    xr[0] = x + (size_t)(row0 + l15) * ND;
    xr[1] = x + (size_t)(row0 + 16 + l15) * ND;

    float sacc[2][8];
#pragma unroll
    for (int m = 0; m < 2; ++m)
#pragma unroll
        for (int r = 0; r < 8; ++r) sacc[m][r] = 0.0f;

    const v8f vzero = {};

    for (int g = 0; g < 8; ++g) {               // 8 groups x 4 n-tiles = 32 tiles
        const int nbase = g * 64;

        __syncthreads();                        // prev group's LDS reads done
        // Stage Wt[nbase..nbase+64) x K=512 into LDS: 32768 halfs,
        // 256 threads x 16 chunks of 8 halfs (16 B) each.
#pragma unroll
        for (int it = 0; it < 16; ++it) {
            int idx = it * 256 + tid;           // 0..4095 chunks
            int nl  = idx >> 6;                 // 64 chunks per column
            int kk  = (idx & 63) * 8;
            const _Float16* src = Wt + (size_t)(nbase + nl) * ND + kk;
            _Float16*       dst = &Bsh[nl * LDSLD + kk];
#if USE_ASYNC_LDS
            __builtin_amdgcn_global_load_async_to_lds_b128(
                (gv4i*)(src), (lv4i*)(dst), 0, 0);
#else
            *(v8h*)dst = *(const v8h*)src;
#endif
        }
#if USE_ASYNC_LDS
        __builtin_amdgcn_s_wait_asynccnt(0);
#endif
        __syncthreads();

        v8f acc[2][4];
#pragma unroll
        for (int m = 0; m < 2; ++m)
#pragma unroll
            for (int j = 0; j < 4; ++j) acc[m][j] = vzero;

        for (int kt = 0; kt < 16; ++kt) {
            const int k0 = kt * 32;

            // B fragments from LDS (layout: lanes 0-15 K[k0,k0+8)+[k0+16,k0+24),
            // lanes 16-31 K[k0+8,k0+16)+[k0+24,k0+32); col = j*16+l15).
            v16h bf[4];
#pragma unroll
            for (int j = 0; j < 4; ++j) {
                const _Float16* br = &Bsh[(j * 16 + l15) * LDSLD];
                v8h b0 = *(const v8h*)(br + k0 + hs * 8);
                v8h b1 = *(const v8h*)(br + k0 + 16 + hs * 8);
#pragma unroll
                for (int i = 0; i < 8; ++i) { bf[j][i] = b0[i]; bf[j][8 + i] = b1[i]; }
            }

#pragma unroll
            for (int m = 0; m < 2; ++m) {
                v16h a;
                const float* pa0 = xr[m] + k0 + hs * 8;
                const float* pa1 = xr[m] + k0 + 16 + hs * 8;
#pragma unroll
                for (int i = 0; i < 8; ++i) a[i]     = (_Float16)pa0[i];
#pragma unroll
                for (int i = 0; i < 8; ++i) a[8 + i] = (_Float16)pa1[i];

#pragma unroll
                for (int j = 0; j < 4; ++j) {
                    acc[m][j] = __builtin_amdgcn_wmma_f32_16x16x32_f16(
                        false, a, false, bf[j], (short)0, acc[m][j], false, false);
                }
            }
        }

        // C tile: acc[m][j][r] = (row = r + 8*hs, col = l15) of n-tile j.
#pragma unroll
        for (int j = 0; j < 4; ++j) {
            const float vv = attv[nbase + j * 16 + l15];
#pragma unroll
            for (int m = 0; m < 2; ++m)
#pragma unroll
                for (int r = 0; r < 8; ++r)
                    sacc[m][r] += fast_tanh(acc[m][j][r]) * vv;
        }
    }

    // Reduce across the 16 lanes of each half-wave (cols) -> per-row score.
#pragma unroll
    for (int m = 0; m < 2; ++m) {
#pragma unroll
        for (int r = 0; r < 8; ++r) {
            float s = sacc[m][r];
            s += __shfl_xor(s, 1, 32);
            s += __shfl_xor(s, 2, 32);
            s += __shfl_xor(s, 4, 32);
            s += __shfl_xor(s, 8, 32);
            sacc[m][r] = s;
        }
        if (l15 == 0) {
#pragma unroll
            for (int r = 0; r < 8; ++r)
                scores[row0 + m * 16 + hs * 8 + r] = sacc[m][r];
        }
    }
}

// ---------------------------------------------------------------------------
// Kernel 3: softmax over T per batch
// ---------------------------------------------------------------------------
__global__ void __launch_bounds__(256)
softmax_kernel(const float* __restrict__ scores, float* __restrict__ weights) {
    const int b   = blockIdx.x;
    const int tid = threadIdx.x;
    __shared__ float red[256];
    const float* s = scores + (size_t)b * NT;

    float lmax = -INFINITY;
    for (int t = tid; t < NT; t += 256) lmax = fmaxf(lmax, s[t]);
    red[tid] = lmax;
    __syncthreads();
    for (int off = 128; off > 0; off >>= 1) {
        if (tid < off) red[tid] = fmaxf(red[tid], red[tid + off]);
        __syncthreads();
    }
    const float m = red[0];
    __syncthreads();

    float lsum = 0.0f;
    for (int t = tid; t < NT; t += 256) lsum += __expf(s[t] - m);
    red[tid] = lsum;
    __syncthreads();
    for (int off = 128; off > 0; off >>= 1) {
        if (tid < off) red[tid] += red[tid + off];
        __syncthreads();
    }
    const float inv = 1.0f / red[0];

    float* w = weights + (size_t)b * NT;
    for (int t = tid; t < NT; t += 256) w[t] = __expf(s[t] - m) * inv;
}

// ---------------------------------------------------------------------------
// Kernel 4: out[b][d] = sum_t w[b][t] * x[b][t][d]   (x is L2-resident)
// ---------------------------------------------------------------------------
__global__ void __launch_bounds__(512)
pool_kernel(const float* __restrict__ x, const float* __restrict__ weights,
            float* __restrict__ out) {
    const int b = blockIdx.x;
    const int d = threadIdx.x;
    const float* w  = weights + (size_t)b * NT;
    const float* xb = x + (size_t)b * NT * ND;
    float a0 = 0.f, a1 = 0.f, a2 = 0.f, a3 = 0.f;
    for (int t = 0; t < NT; t += 4) {
        a0 = fmaf(w[t + 0], xb[(size_t)(t + 0) * ND + d], a0);
        a1 = fmaf(w[t + 1], xb[(size_t)(t + 1) * ND + d], a1);
        a2 = fmaf(w[t + 2], xb[(size_t)(t + 2) * ND + d], a2);
        a3 = fmaf(w[t + 3], xb[(size_t)(t + 3) * ND + d], a3);
    }
    out[b * ND + d] = (a0 + a1) + (a2 + a3);
}

// ---------------------------------------------------------------------------
extern "C" void kernel_launch(void* const* d_in, const int* in_sizes, int n_in,
                              void* d_out, int out_size, void* d_ws, size_t ws_size,
                              hipStream_t stream) {
    const float* x    = (const float*)d_in[0];   // [16,4096,512] f32
    const float* W    = (const float*)d_in[1];   // [512,512] f32
    const float* attv = (const float*)d_in[2];   // [512] f32
    float* out = (float*)d_out;                  // [16,512] f32

    char* ws = (char*)d_ws;
    _Float16* Wt     = (_Float16*)(ws);                            // 512 KB
    float*    scores = (float*)(ws + 512u * 1024u);                // 256 KB
    float*    wgt    = (float*)(ws + 512u * 1024u + 256u * 1024u); // 256 KB

    prep_wt_kernel<<<(ND * ND) / 256, 256, 0, stream>>>(W, Wt);
    score_kernel<<<NROWS / 256, 256, 0, stream>>>(x, Wt, attv, scores);
    softmax_kernel<<<NB, 256, 0, stream>>>(scores, wgt);
    pool_kernel<<<NB, ND, 0, stream>>>(x, wgt, out);
}
